// Model_57492432224756
// MI455X (gfx1250) — matrix operand order repaired and checked by reading, not compiled, not run on hardware
//
#include <hip/hip_runtime.h>
#include <math.h>

#define T_SEQ 2048
#define BATCH 64
#define HID   512                    // H == I == 512, K dim for both GEMMs
#define NBLK  16                     // column-blocks in recurrent kernel
#define CPB   (HID / NBLK)           // 32 columns per block

#define WELEM (HID * HID)            // elements per weight matrix
#define HELEM (BATCH * HID)          // elements per hidden state

typedef __attribute__((ext_vector_type(16))) __bf16 v16bf;
typedef __attribute__((ext_vector_type(8)))  __bf16 v8bf;
typedef __attribute__((ext_vector_type(8)))  float  v8f;
typedef int v4i __attribute__((vector_size(16)));   // matches async-LDS builtin param

#if defined(__gfx1250__) && __has_builtin(__builtin_amdgcn_global_load_async_to_lds_b128)
#define ASYNC_LDS 1
#else
#define ASYNC_LDS 0
#endif

// f32 -> (hi, lo) bf16 limbs; hi = RNE(f), lo = RNE(f - hi)
__device__ __forceinline__ void split1(float f, __bf16& h, __bf16& l) {
  h = (__bf16)f;
  l = (__bf16)(f - (float)h);
}

// C += Ahi*Bhi + Ahi*Blo + Alo*Bhi  (fp32-ish product from bf16 limbs)
__device__ __forceinline__ v8f wmma3(v16bf ahi, v16bf alo, v16bf bhi, v16bf blo, v8f c) {
  c = __builtin_amdgcn_wmma_f32_16x16x32_bf16(false, ahi, false, bhi, (short)0, c, false, false);
  c = __builtin_amdgcn_wmma_f32_16x16x32_bf16(false, ahi, false, blo, (short)0, c, false, false);
  c = __builtin_amdgcn_wmma_f32_16x16x32_bf16(false, alo, false, bhi, (short)0, c, false, false);
  return c;
}

// A fragment (16x32 bf16) from packed bf16 row-major [row][k]:
// lane half=0: K = k0..k0+15 contiguous; half=1: shifted +16.  Two b128 loads.
__device__ __forceinline__ v16bf load_a_bf(const __bf16* __restrict__ base,
                                           int row, int ka) {
  const __bf16* p = base + (size_t)row * HID + ka;
  v8bf a = *(const v8bf*)p;
  v8bf b = *(const v8bf*)(p + 8);
  return __builtin_shufflevector(a, b, 0,1,2,3,4,5,6,7,8,9,10,11,12,13,14,15);
}

// B fragment (32x16 bf16) for C = A*W^T from packed bf16 [col][k]:
// lane half=0: K = kb..kb+15 contiguous.  One 32-byte vector load.
__device__ __forceinline__ v16bf load_b_bf(const __bf16* __restrict__ base,
                                           int col, int kb) {
  return *(const v16bf*)(base + (size_t)col * HID + kb);
}

// ---------------------------------------------------------------------------
// Kernel 0: decompose W_ih and W_hh into bf16 hi/lo limb arrays (once).
// ---------------------------------------------------------------------------
__global__ void __launch_bounds__(256)
convert_w_kernel(const float* __restrict__ Wih, const float* __restrict__ Whh,
                 __bf16* __restrict__ wih_hi, __bf16* __restrict__ wih_lo,
                 __bf16* __restrict__ whh_hi, __bf16* __restrict__ whh_lo) {
  int i = blockIdx.x * 256 + threadIdx.x;          // 0 .. 2*WELEM-1
  float f = (i < WELEM) ? Wih[i] : Whh[i - WELEM];
  __bf16 h, l;
  split1(f, h, l);
  if (i < WELEM) { wih_hi[i] = h; wih_lo[i] = l; }
  else           { whh_hi[i - WELEM] = h; whh_lo[i - WELEM] = l; }
}

// ---------------------------------------------------------------------------
// Kernel 1: xw[r, n] = sum_k x[r,k]*W_ih[n,k] + b_ih[n] + b_hh[n]
// Block 256 threads = 8 waves; block tile 128(M) x 64(N); wave tile 16 x 64.
// ---------------------------------------------------------------------------
__global__ void __launch_bounds__(256)
xw_gemm_kernel(const float* __restrict__ x,
               const __bf16* __restrict__ wih_hi, const __bf16* __restrict__ wih_lo,
               const float* __restrict__ bih, const float* __restrict__ bhh,
               float* __restrict__ out) {
  const int lane = threadIdx.x & 31;
  const int wave = threadIdx.x >> 5;
  const int half = lane >> 4;
  const int l15  = lane & 15;
  const int row0 = blockIdx.x * 128 + wave * 16;
  const int col0 = blockIdx.y * 64;

  const float* xr = x + (size_t)(row0 + l15) * HID;

  v8f acc[4] = {};
  for (int k0 = 0; k0 < HID; k0 += 32) {
    const int ka = k0 + half * 16;
    float t[16];
#pragma unroll
    for (int i = 0; i < 8; ++i) t[i]     = xr[ka + i];
#pragma unroll
    for (int i = 0; i < 8; ++i) t[8 + i] = xr[ka + 8 + i];
    v16bf ahi, alo;
#pragma unroll
    for (int i = 0; i < 16; ++i) {
      __bf16 h, l;
      split1(t[i], h, l);
      ahi[i] = h; alo[i] = l;
    }
    const int kb = k0 + half * 16;
#pragma unroll
    for (int nt = 0; nt < 4; ++nt) {
      v16bf bhi = load_b_bf(wih_hi, col0 + nt * 16 + l15, kb);
      v16bf blo = load_b_bf(wih_lo, col0 + nt * 16 + l15, kb);
      acc[nt] = wmma3(ahi, alo, bhi, blo, acc[nt]);
    }
  }
#pragma unroll
  for (int nt = 0; nt < 4; ++nt) {
    const int col = col0 + nt * 16 + l15;
    const float bias = bih[col] + bhh[col];
#pragma unroll
    for (int j = 0; j < 8; ++j) {
      const int row = row0 + half * 8 + j;   // C layout: VGPR j -> M = j + 8*half
      out[(size_t)row * HID + col] = acc[nt][j] + bias;
    }
  }
}

// ---------------------------------------------------------------------------
// Kernel 2: h_0 = tanh(xw_0) in place + emit h_0 limbs; zero the grid-sync ctr.
// ---------------------------------------------------------------------------
__global__ void __launch_bounds__(256)
h0_kernel(float* __restrict__ p, __bf16* __restrict__ hp,
          unsigned* __restrict__ sync_counter) {
  int i = blockIdx.x * 256 + threadIdx.x;          // 0 .. HELEM-1
  if (i == 0) *sync_counter = 0u;
  float v = tanhf(p[i]);
  p[i] = v;
  __bf16 h, l;
  split1(v, h, l);
  hp[i] = h;
  hp[HELEM + i] = l;
}

// ---------------------------------------------------------------------------
// Kernel 3: persistent recurrent kernel — ONE launch for all 2047 steps.
// Grid: NBLK=16 blocks (32 cols each); block 128 threads = 4 waves x 16 rows.
// Each block stages its W_hh limb slice (64 KB) into LDS once via async
// global->LDS, then loops t with an agent-scope release/acquire barrier.
// ---------------------------------------------------------------------------
__global__ void __launch_bounds__(128)
rnn_persistent_kernel(const __bf16* __restrict__ whh_hi,
                      const __bf16* __restrict__ whh_lo,
                      float* __restrict__ out,          // [T*B*H], xw -> h in place
                      __bf16* __restrict__ hbuf,        // [2 ping][2 limb][HELEM]
                      unsigned* __restrict__ sync_counter) {
  __shared__ __bf16 sw[2][CPB * HID];                   // 64 KB: hi, lo limb slices

  const int col0 = blockIdx.x * CPB;

  // ---- one-time stage: W_hh cols [col0, col0+CPB) limbs -> LDS (16B units) ----
  const int UNITS = 2 * CPB * HID / 8;                  // 4096 units of 8 bf16
  for (int u = threadIdx.x; u < UNITS; u += 128) {
    const int limb = u >> 11;                           // 2048 units per limb
    const int o8   = (u & 2047) * 8;
    const __bf16* gsrc = (limb ? whh_lo : whh_hi) + (size_t)col0 * HID + o8;
    __bf16* ldst = &sw[limb][o8];
#if ASYNC_LDS
    __builtin_amdgcn_global_load_async_to_lds_b128(
        (__attribute__((address_space(1))) v4i*)gsrc,
        (__attribute__((address_space(3))) v4i*)ldst, 0, 0);
#else
    *(v8bf*)ldst = *(const v8bf*)gsrc;
#endif
  }
#if ASYNC_LDS
#if __has_builtin(__builtin_amdgcn_s_wait_asynccnt)
  __builtin_amdgcn_s_wait_asynccnt(0);
#else
  asm volatile("s_wait_asynccnt 0x0" ::: "memory");
#endif
#endif
  __syncthreads();

  const int lane = threadIdx.x & 31;
  const int wave = threadIdx.x >> 5;
  const int half = lane >> 4;
  const int l15  = lane & 15;
  const int row0 = wave * 16;                           // BATCH = 64 = 4 waves * 16

  for (int t = 1; t < T_SEQ; ++t) {
    const __bf16* hp_hi = hbuf + (size_t)((t - 1) & 1) * (2 * HELEM);
    const __bf16* hp_lo = hp_hi + HELEM;
    __bf16* hp_cur = hbuf + (size_t)(t & 1) * (2 * HELEM);
    float* hxw = out + (size_t)t * HELEM;

    v8f acc[2] = {};
    for (int k0 = 0; k0 < HID; k0 += 32) {
      const int ka = k0 + half * 16;
      v16bf ahi = load_a_bf(hp_hi, row0 + l15, ka);
      v16bf alo = load_a_bf(hp_lo, row0 + l15, ka);
      const int kb = k0 + half * 16;
#pragma unroll
      for (int nt = 0; nt < 2; ++nt) {
        const int cl = nt * 16 + l15;                   // local column
        v16bf bhi = *(const v16bf*)&sw[0][cl * HID + kb];   // ds_load_b128 x2
        v16bf blo = *(const v16bf*)&sw[1][cl * HID + kb];
        acc[nt] = wmma3(ahi, alo, bhi, blo, acc[nt]);
      }
    }
    // epilogue: h = tanh(acc + xw); store f32 result + bf16 limbs for next step
#pragma unroll
    for (int nt = 0; nt < 2; ++nt) {
      const int col = col0 + nt * 16 + l15;
#pragma unroll
      for (int j = 0; j < 8; ++j) {
        const int row = row0 + half * 8 + j;
        const size_t idx = (size_t)row * HID + col;
        float v = tanhf(acc[nt][j] + hxw[idx]);
        hxw[idx] = v;
        __bf16 h, l;
        split1(v, h, l);
        hp_cur[idx] = h;
        hp_cur[HELEM + idx] = l;
      }
    }
    // ---- grid barrier: monotonic counter, release on arrive / acquire on spin ----
    __syncthreads();
    if (threadIdx.x == 0) {
      __hip_atomic_fetch_add(sync_counter, 1u, __ATOMIC_RELEASE,
                             __HIP_MEMORY_SCOPE_AGENT);
      const unsigned target = (unsigned)(NBLK * t);
      while (__hip_atomic_load(sync_counter, __ATOMIC_ACQUIRE,
                               __HIP_MEMORY_SCOPE_AGENT) < target)
        __builtin_amdgcn_s_sleep(2);
    }
    __syncthreads();
  }
}

extern "C" void kernel_launch(void* const* d_in, const int* in_sizes, int n_in,
                              void* d_out, int out_size, void* d_ws, size_t ws_size,
                              hipStream_t stream) {
  (void)in_sizes; (void)n_in; (void)out_size; (void)ws_size;
  const float* x   = (const float*)d_in[0];   // [T, B, I]
  const float* Wih = (const float*)d_in[1];   // [H, I]
  const float* Whh = (const float*)d_in[2];   // [H, H]
  const float* bih = (const float*)d_in[3];   // [H]
  const float* bhh = (const float*)d_in[4];   // [H]
  float* out = (float*)d_out;                 // [T*B, H]

  // Workspace carve (~2.25 MB): W limbs + h limb ping-pong + sync counter.
  __bf16* wih_hi = (__bf16*)d_ws;
  __bf16* wih_lo = wih_hi + WELEM;
  __bf16* whh_hi = wih_lo + WELEM;
  __bf16* whh_lo = whh_hi + WELEM;
  __bf16* hbuf   = whh_lo + WELEM;            // [2][2][HELEM]
  unsigned* ctr  = (unsigned*)(hbuf + 4 * HELEM);

  // 0) decompose weights into bf16 hi/lo limbs
  convert_w_kernel<<<(2 * WELEM) / 256, 256, 0, stream>>>(
      Wih, Whh, wih_hi, wih_lo, whh_hi, whh_lo);

  // 1) xw = x @ W_ih^T + b_ih + b_hh  -> directly into d_out
  dim3 g1((T_SEQ * BATCH) / 128, HID / 64);
  xw_gemm_kernel<<<g1, 256, 0, stream>>>(x, wih_hi, wih_lo, bih, bhh, out);

  // 2) h_0 = tanh(xw_0); emit limbs into ping 0; zero grid-sync counter
  h0_kernel<<<HELEM / 256, 256, 0, stream>>>(out, hbuf, ctr);

  // 3) one persistent launch for the whole recurrence
  rnn_persistent_kernel<<<NBLK, 128, 0, stream>>>(whh_hi, whh_lo, out, hbuf, ctr);
}
